// LowRank_35862976921797
// MI455X (gfx1250) — matrix operand order repaired and checked by reading, not compiled, not run on hardware
//
#include <hip/hip_runtime.h>
#include <hip/hip_bf16.h>
#include <stdint.h>

// ---------------------------------------------------------------------------
// MI455X / gfx1250, wave32, WMMA bf16 path.
// Shapes: B=32, M=1024, E=1024, H=8, D=128, MID=64
// ---------------------------------------------------------------------------

typedef __attribute__((ext_vector_type(16))) __bf16 v16bf;
typedef __attribute__((ext_vector_type(8)))  float  v8f;

#define E_DIM   1024
#define B_DIM   32
#define M_DIM   1024
#define H_DIM   8
#define D_DIM   128
#define MID_DIM 64
#define GN_EPS  1e-5f

__device__ __forceinline__ unsigned short f2bf(float x) {
    __bf16 h = (__bf16)x;
    return __builtin_bit_cast(unsigned short, h);
}
__device__ __forceinline__ float bflo(unsigned u) { return __uint_as_float(u << 16); }
__device__ __forceinline__ float bfhi(unsigned u) { return __uint_as_float(u & 0xffff0000u); }

// A-matrix bf16 16x32 layout: lane holds row m=l%16; VGPR j covers
// k0 = 16*(j/4) + 8*(l/16) + 2*(j%4), packed (k0, k0+1).
__device__ __forceinline__ int a_k0(int j, int half) {
    return 16 * (j >> 2) + 8 * half + 2 * (j & 3);
}

// ---------------------------------------------------------------------------
// Pack a row-major f32 weight W[K][N] into WMMA-B bf16 register layout:
// P[((kt*NT+nt)*32 + lane)*8 + j] = pack(W[k][n], W[k+1][n]),
//   n = nt*16 + lane%16,  k = kt*32 + 16*(j/4) + 8*(lane/16) + 2*(j%4)
// ---------------------------------------------------------------------------
__global__ void pack_w_kernel(const float* __restrict__ W, unsigned* __restrict__ P,
                              int KT, int NT, int N) {
    int idx = blockIdx.x * blockDim.x + threadIdx.x;
    int total = KT * NT * 256;
    if (idx >= total) return;
    int j = idx & 7;
    int lane = (idx >> 3) & 31;
    int tile = idx >> 8;
    int nt = tile % NT;
    int kt = tile / NT;
    int n = nt * 16 + (lane & 15);
    int k = kt * 32 + 16 * (j >> 2) + 8 * (lane >> 4) + 2 * (j & 3);
    float f0 = W[(size_t)k * N + n];
    float f1 = W[(size_t)(k + 1) * N + n];
    P[idx] = ((unsigned)f2bf(f1) << 16) | (unsigned)f2bf(f0);
}

// ---------------------------------------------------------------------------
// Fused Linear(E,E) + bias + ReLU + GroupNorm(H groups).
// One block (8 waves) per 32-row super-tile (2 row tiles register-blocked so
// each packed-B tile is reused twice -> halves B-side L2 traffic).
// Wave w owns column group w (128 ch) == one GroupNorm group -> wave-local
// stats via shfl_xor over the 16-lane C-matrix halves.
// ---------------------------------------------------------------------------
__global__ __launch_bounds__(256) void proj_gn_kernel(
    const float* __restrict__ X, const unsigned* __restrict__ Wp,
    const float* __restrict__ bias, const float* __restrict__ gamma,
    const float* __restrict__ beta,
    float* __restrict__ outF, unsigned short* __restrict__ outB, int outIsBF16) {
    const int tid = threadIdx.x, w = tid >> 5, l = tid & 31;
    const int n = l & 15, half = l >> 4;
    const long rows0 = (long)blockIdx.x * 32;
    const int NT = E_DIM / 16;

    v8f acc[2][8];
#pragma unroll
    for (int rr = 0; rr < 2; ++rr)
#pragma unroll
        for (int t = 0; t < 8; ++t)
#pragma unroll
            for (int r = 0; r < 8; ++r) acc[rr][t][r] = 0.f;

    const float* arow0 = X + (rows0 + n) * E_DIM;       // row tile 0, row = lane%16
    const float* arow1 = X + (rows0 + 16 + n) * E_DIM;  // row tile 1
    for (int kt = 0; kt < E_DIM / 32; ++kt) {
        v16bf a0, a1;
        __builtin_prefetch(arow0 + (kt + 4) * 32, 0, 1);
        __builtin_prefetch(arow1 + (kt + 4) * 32, 0, 1);
#pragma unroll
        for (int j = 0; j < 8; ++j) {
            int k = kt * 32 + a_k0(j, half);
            float2 f0 = *reinterpret_cast<const float2*>(arow0 + k);
            float2 f1 = *reinterpret_cast<const float2*>(arow1 + k);
            a0[2 * j] = (__bf16)f0.x; a0[2 * j + 1] = (__bf16)f0.y;
            a1[2 * j] = (__bf16)f1.x; a1[2 * j + 1] = (__bf16)f1.y;
        }
#pragma unroll
        for (int t = 0; t < 8; ++t) {
            int nt = w * 8 + t;
            v16bf bv = *reinterpret_cast<const v16bf*>(Wp + ((size_t)(kt * NT + nt) * 32 + l) * 8);
            acc[0][t] = __builtin_amdgcn_wmma_f32_16x16x32_bf16(
                false, a0, false, bv, (short)0, acc[0][t], false, false);
            acc[1][t] = __builtin_amdgcn_wmma_f32_16x16x32_bf16(
                false, a1, false, bv, (short)0, acc[1][t], false, false);
        }
    }

    // bias + relu + GroupNorm per row tile
#pragma unroll
    for (int rr = 0; rr < 2; ++rr) {
        float s[8], s2[8];
#pragma unroll
        for (int r = 0; r < 8; ++r) { s[r] = 0.f; s2[r] = 0.f; }
#pragma unroll
        for (int t = 0; t < 8; ++t) {
            int c = w * 128 + t * 16 + n;
            float bc = bias[c];
#pragma unroll
            for (int r = 0; r < 8; ++r) {
                float v = acc[rr][t][r] + bc;
                v = v > 0.f ? v : 0.f;
                acc[rr][t][r] = v;
                s[r] += v;
                s2[r] += v * v;
            }
        }
#pragma unroll
        for (int r = 0; r < 8; ++r) {
#pragma unroll
            for (int off = 1; off < 16; off <<= 1) {
                s[r]  += __shfl_xor(s[r], off);
                s2[r] += __shfl_xor(s2[r], off);
            }
        }
        float mean[8], inv[8];
#pragma unroll
        for (int r = 0; r < 8; ++r) {
            mean[r] = s[r] * (1.f / 128.f);
            float var = s2[r] * (1.f / 128.f) - mean[r] * mean[r];
            inv[r] = rsqrtf(var + GN_EPS);
        }
#pragma unroll
        for (int t = 0; t < 8; ++t) {
            int c = w * 128 + t * 16 + n;
            float g = gamma[c], be = beta[c];
#pragma unroll
            for (int r = 0; r < 8; ++r) {
                float y = (acc[rr][t][r] - mean[r]) * inv[r] * g + be;
                long row = rows0 + rr * 16 + r + 8 * half;
                if (outIsBF16) outB[row * E_DIM + c] = f2bf(y);
                else           outF[row * E_DIM + c] = y;
            }
        }
    }
}

// ---------------------------------------------------------------------------
// Score kernel: h = relu((q \odot k) @ W1 + b1) computed tile-by-tile;
// emits logits = h @ Wls + bls and masked pool mean, never storing h.
// One block per (b,h); 8 waves x 8 row-tiles each cover M=1024.
// The 16 packed-B tiles of W1 (512 B/lane) are register-cached per wave.
// ---------------------------------------------------------------------------
__global__ __launch_bounds__(256) void score_kernel(
    const float* __restrict__ qn, const unsigned short* __restrict__ kn,
    const unsigned* __restrict__ W1p, const float* __restrict__ b1,
    const float* __restrict__ Wls, const float* __restrict__ bls,
    const int* __restrict__ mask,
    float* __restrict__ logits, float* __restrict__ pool) {
    const int bh = blockIdx.x;
    const int b = bh >> 3, h = bh & 7;
    const int tid = threadIdx.x, w = tid >> 5, l = tid & 31;
    const int n = l & 15, half = l >> 4;

    __shared__ float poolAcc[MID_DIM];
    __shared__ int msum;
    if (tid < MID_DIM) poolAcc[tid] = 0.f;
    if (tid == 0) msum = 0;
    __syncthreads();
    int lm = 0;
    for (int m = tid; m < M_DIM; m += 256) lm += mask[b * M_DIM + m];
    atomicAdd(&msum, lm);

    const int qbase = b * E_DIM + h * D_DIM;
    float wlsv[4];
#pragma unroll
    for (int t = 0; t < 4; ++t) wlsv[t] = Wls[t * 16 + n];

    // register-cache all 16 B tiles of packed W1
    v16bf bcache[16];
#pragma unroll
    for (int kt = 0; kt < 4; ++kt)
#pragma unroll
        for (int t = 0; t < 4; ++t)
            bcache[kt * 4 + t] =
                *reinterpret_cast<const v16bf*>(W1p + ((size_t)(kt * 4 + t) * 32 + l) * 8);

    for (int i = 0; i < 8; ++i) {
        const int m0 = (w * 8 + i) * 16;
        v8f acc[4];
#pragma unroll
        for (int t = 0; t < 4; ++t)
#pragma unroll
            for (int r = 0; r < 8; ++r) acc[t][r] = 0.f;

        const size_t kbase = ((size_t)(b * M_DIM + m0 + n)) * E_DIM + h * D_DIM;
        __builtin_prefetch(kn + kbase + 16 * E_DIM, 0, 1);
#pragma unroll
        for (int kt = 0; kt < 4; ++kt) {
            v16bf a;
#pragma unroll
            for (int j = 0; j < 8; ++j) {
                int k = kt * 32 + a_k0(j, half);
                unsigned u = *reinterpret_cast<const unsigned*>(kn + kbase + k);
                float klo = bflo(u), khi = bfhi(u);
                float q0 = qn[qbase + k], q1 = qn[qbase + k + 1];
                a[2 * j]     = (__bf16)(klo * q0);
                a[2 * j + 1] = (__bf16)(khi * q1);
            }
#pragma unroll
            for (int t = 0; t < 4; ++t) {
                acc[t] = __builtin_amdgcn_wmma_f32_16x16x32_bf16(
                    false, a, false, bcache[kt * 4 + t], (short)0, acc[t], false, false);
            }
        }
        // bias + relu
        float mrow[8];
#pragma unroll
        for (int r = 0; r < 8; ++r) mrow[r] = (float)mask[b * M_DIM + m0 + r + 8 * half];
#pragma unroll
        for (int t = 0; t < 4; ++t) {
            float bc = b1[t * 16 + n];
#pragma unroll
            for (int r = 0; r < 8; ++r) {
                float v = acc[t][r] + bc;
                acc[t][r] = v > 0.f ? v : 0.f;
            }
        }
        // logits: per-row dot with Wls, reduce over 16-lane half
#pragma unroll
        for (int r = 0; r < 8; ++r) {
            float ls = 0.f;
#pragma unroll
            for (int t = 0; t < 4; ++t) ls += acc[t][r] * wlsv[t];
#pragma unroll
            for (int off = 1; off < 16; off <<= 1) ls += __shfl_xor(ls, off);
            if (n == 0)
                logits[(size_t)bh * M_DIM + m0 + r + 8 * half] = ls + bls[0];
        }
        // masked pool sums
#pragma unroll
        for (int t = 0; t < 4; ++t) {
            float p = 0.f;
#pragma unroll
            for (int r = 0; r < 8; ++r) p += acc[t][r] * mrow[r];
            p += __shfl_xor(p, 16);
            if (half == 0) atomicAdd(&poolAcc[t * 16 + n], p);
        }
    }
    __syncthreads();
    if (tid < MID_DIM) {
        float cnt = fmaxf((float)msum, 1.f);
        pool[(size_t)bh * MID_DIM + tid] = poolAcc[tid] / cnt;
    }
}

// ---------------------------------------------------------------------------
// Masked softmax over M per (b,h).
// ---------------------------------------------------------------------------
__global__ __launch_bounds__(128) void softmax_kernel(
    const float* __restrict__ logits, const int* __restrict__ mask,
    float* __restrict__ alpha) {
    const int bh = blockIdx.x, b = bh >> 3;
    const int tid = threadIdx.x;
    __shared__ float red[128];
    float x[8];
#pragma unroll
    for (int i = 0; i < 8; ++i) {
        int m = tid + 128 * i;
        int mk = mask[b * M_DIM + m];
        x[i] = mk ? logits[(size_t)bh * M_DIM + m] : -1e9f;
    }
    float mx = x[0];
#pragma unroll
    for (int i = 1; i < 8; ++i) mx = fmaxf(mx, x[i]);
    red[tid] = mx; __syncthreads();
    for (int s = 64; s > 0; s >>= 1) {
        if (tid < s) red[tid] = fmaxf(red[tid], red[tid + s]);
        __syncthreads();
    }
    mx = red[0]; __syncthreads();
    float e[8], lsum = 0.f;
#pragma unroll
    for (int i = 0; i < 8; ++i) { e[i] = __expf(x[i] - mx); lsum += e[i]; }
    red[tid] = lsum; __syncthreads();
    for (int s = 64; s > 0; s >>= 1) {
        if (tid < s) red[tid] += red[tid + s];
        __syncthreads();
    }
    float rtot = 1.f / red[0];
#pragma unroll
    for (int i = 0; i < 8; ++i)
        alpha[(size_t)bh * M_DIM + tid + 128 * i] = e[i] * rtot;
}

// ---------------------------------------------------------------------------
// Fused v2 projection + GroupNorm + alpha-weighted aggregation:
// v2_agg[b, h*128+d] += sum over two 16-row tiles of alpha * v2n.
// Avoids materializing the 128 MB projected v2 tensor.
// ---------------------------------------------------------------------------
__global__ __launch_bounds__(256) void v2_proj_agg_kernel(
    const float* __restrict__ X, const unsigned* __restrict__ Wp,
    const float* __restrict__ bias, const float* __restrict__ gamma,
    const float* __restrict__ beta,
    const float* __restrict__ alpha, float* __restrict__ v2agg) {
    const int tid = threadIdx.x, w = tid >> 5, l = tid & 31;
    const int n = l & 15, half = l >> 4;
    const long rows0 = (long)blockIdx.x * 32;
    const int b = (int)(blockIdx.x >> 5);
    const int m00 = ((int)blockIdx.x & 31) * 32;
    const int NT = E_DIM / 16;

    v8f acc[2][8];
#pragma unroll
    for (int rr = 0; rr < 2; ++rr)
#pragma unroll
        for (int t = 0; t < 8; ++t)
#pragma unroll
            for (int r = 0; r < 8; ++r) acc[rr][t][r] = 0.f;

    const float* arow0 = X + (rows0 + n) * E_DIM;
    const float* arow1 = X + (rows0 + 16 + n) * E_DIM;
    for (int kt = 0; kt < E_DIM / 32; ++kt) {
        v16bf a0, a1;
        __builtin_prefetch(arow0 + (kt + 4) * 32, 0, 1);
        __builtin_prefetch(arow1 + (kt + 4) * 32, 0, 1);
#pragma unroll
        for (int j = 0; j < 8; ++j) {
            int k = kt * 32 + a_k0(j, half);
            float2 f0 = *reinterpret_cast<const float2*>(arow0 + k);
            float2 f1 = *reinterpret_cast<const float2*>(arow1 + k);
            a0[2 * j] = (__bf16)f0.x; a0[2 * j + 1] = (__bf16)f0.y;
            a1[2 * j] = (__bf16)f1.x; a1[2 * j + 1] = (__bf16)f1.y;
        }
#pragma unroll
        for (int t = 0; t < 8; ++t) {
            int nt = w * 8 + t;
            v16bf bv = *reinterpret_cast<const v16bf*>(Wp + ((size_t)(kt * NT + nt) * 32 + l) * 8);
            acc[0][t] = __builtin_amdgcn_wmma_f32_16x16x32_bf16(
                false, a0, false, bv, (short)0, acc[0][t], false, false);
            acc[1][t] = __builtin_amdgcn_wmma_f32_16x16x32_bf16(
                false, a1, false, bv, (short)0, acc[1][t], false, false);
        }
    }

#pragma unroll
    for (int rr = 0; rr < 2; ++rr) {
        const int m0 = m00 + rr * 16;
        float s[8], s2[8];
#pragma unroll
        for (int r = 0; r < 8; ++r) { s[r] = 0.f; s2[r] = 0.f; }
#pragma unroll
        for (int t = 0; t < 8; ++t) {
            int c = w * 128 + t * 16 + n;
            float bc = bias[c];
#pragma unroll
            for (int r = 0; r < 8; ++r) {
                float v = acc[rr][t][r] + bc;
                v = v > 0.f ? v : 0.f;
                acc[rr][t][r] = v;
                s[r] += v;
                s2[r] += v * v;
            }
        }
#pragma unroll
        for (int r = 0; r < 8; ++r) {
#pragma unroll
            for (int off = 1; off < 16; off <<= 1) {
                s[r]  += __shfl_xor(s[r], off);
                s2[r] += __shfl_xor(s2[r], off);
            }
        }
        float mean[8], inv[8], alf[8];
#pragma unroll
        for (int r = 0; r < 8; ++r) {
            mean[r] = s[r] * (1.f / 128.f);
            float var = s2[r] * (1.f / 128.f) - mean[r] * mean[r];
            inv[r] = rsqrtf(var + GN_EPS);
            // wave w == head w; alpha for this lane-half's rows
            alf[r] = alpha[((size_t)(b * H_DIM + w)) * M_DIM + m0 + r + 8 * half];
        }
#pragma unroll
        for (int t = 0; t < 8; ++t) {
            int c = w * 128 + t * 16 + n;
            float g = gamma[c], be = beta[c];
            float part = 0.f;
#pragma unroll
            for (int r = 0; r < 8; ++r) {
                float y = (acc[rr][t][r] - mean[r]) * inv[r] * g + be;
                part += y * alf[r];
            }
            part += __shfl_xor(part, 16);  // combine rows 0-7 and 8-15
            if (half == 0) atomicAdd(&v2agg[b * E_DIM + c], part);
        }
    }
}

// ---------------------------------------------------------------------------
// Final: alpha_c = sigmoid(pool @ Wlc + blc); out = v1 * v2_agg * alpha_c
// ---------------------------------------------------------------------------
__global__ __launch_bounds__(256) void final_kernel(
    const float* __restrict__ v1n, const float* __restrict__ v2agg,
    const float* __restrict__ pool, const float* __restrict__ Wlc,
    const float* __restrict__ blc, float* __restrict__ out) {
    int idx = blockIdx.x * blockDim.x + threadIdx.x;
    if (idx >= B_DIM * E_DIM) return;
    int b = idx >> 10, c = idx & 1023;
    int h = c >> 7, d = c & 127;
    const float* pl = pool + (size_t)(b * H_DIM + h) * MID_DIM;
    float g = blc[d];
#pragma unroll 8
    for (int mid = 0; mid < MID_DIM; ++mid) g += pl[mid] * Wlc[mid * D_DIM + d];
    g = 1.f / (1.f + __expf(-g));
    out[idx] = v1n[idx] * v2agg[idx] * g;
}

// ---------------------------------------------------------------------------
extern "C" void kernel_launch(void* const* d_in, const int* in_sizes, int n_in,
                              void* d_out, int out_size, void* d_ws, size_t ws_size,
                              hipStream_t stream) {
    (void)in_sizes; (void)n_in; (void)out_size;
    const float* query  = (const float*)d_in[0];
    const float* key    = (const float*)d_in[1];
    const int*   mask   = (const int*)  d_in[2];
    const float* value1 = (const float*)d_in[3];
    const float* value2 = (const float*)d_in[4];
    const float* Wq = (const float*)d_in[5],  *bq = (const float*)d_in[6],
               * gq = (const float*)d_in[7],  *cq = (const float*)d_in[8];
    const float* Wk = (const float*)d_in[9],  *bk = (const float*)d_in[10],
               * gk = (const float*)d_in[11], *ck = (const float*)d_in[12];
    const float* Wv1 = (const float*)d_in[13], *bv1 = (const float*)d_in[14],
               * gv1 = (const float*)d_in[15], *cv1 = (const float*)d_in[16];
    const float* Wv2 = (const float*)d_in[17], *bv2 = (const float*)d_in[18],
               * gv2 = (const float*)d_in[19], *cv2 = (const float*)d_in[20];
    const float* W1  = (const float*)d_in[21], *b1  = (const float*)d_in[22];
    const float* Wls = (const float*)d_in[23], *bls = (const float*)d_in[24];
    const float* Wlc = (const float*)d_in[25], *blc = (const float*)d_in[26];
    float* out = (float*)d_out;

    // workspace carve-up (256B aligned)
    uint8_t* ws = (uint8_t*)d_ws;
    size_t off = 0;
    auto take = [&](size_t bytes) -> uint8_t* {
        uint8_t* p = ws + off;
        off = (off + bytes + 255) & ~(size_t)255;
        return p;
    };
    const size_t PACK_E = (size_t)(E_DIM / 32) * (E_DIM / 16) * 256 * 4;  // 2 MB
    unsigned* pWq  = (unsigned*)take(PACK_E);
    unsigned* pWk  = (unsigned*)take(PACK_E);
    unsigned* pWv1 = (unsigned*)take(PACK_E);
    unsigned* pWv2 = (unsigned*)take(PACK_E);
    unsigned* pW1  = (unsigned*)take((size_t)4 * 4 * 256 * 4);            // 16 KB
    float* qn      = (float*)take((size_t)B_DIM * E_DIM * 4);
    float* v1n     = (float*)take((size_t)B_DIM * E_DIM * 4);
    unsigned short* kn = (unsigned short*)take((size_t)B_DIM * M_DIM * E_DIM * 2);  // 64 MB
    float* logits  = (float*)take((size_t)B_DIM * H_DIM * M_DIM * 4);
    float* pool    = (float*)take((size_t)B_DIM * H_DIM * MID_DIM * 4);
    float* alpha   = (float*)take((size_t)B_DIM * H_DIM * M_DIM * 4);
    float* v2agg   = (float*)take((size_t)B_DIM * E_DIM * 4);
    if (off > ws_size) return;  // scratch too small — bail deterministically

    // 1) pack weights into WMMA-B bf16 layout
    {
        int totE = (E_DIM / 32) * (E_DIM / 16) * 256;  // 524288
        dim3 g((totE + 255) / 256), blk(256);
        pack_w_kernel<<<g, blk, 0, stream>>>(Wq,  pWq,  E_DIM / 32, E_DIM / 16, E_DIM);
        pack_w_kernel<<<g, blk, 0, stream>>>(Wk,  pWk,  E_DIM / 32, E_DIM / 16, E_DIM);
        pack_w_kernel<<<g, blk, 0, stream>>>(Wv1, pWv1, E_DIM / 32, E_DIM / 16, E_DIM);
        pack_w_kernel<<<g, blk, 0, stream>>>(Wv2, pWv2, E_DIM / 32, E_DIM / 16, E_DIM);
        int tot1 = 4 * 4 * 256;
        pack_w_kernel<<<dim3((tot1 + 255) / 256), blk, 0, stream>>>(W1, pW1, 4, 4, MID_DIM);
    }
    // 2) projections (fused GEMM + bias + ReLU + GroupNorm), 32 rows per block
    proj_gn_kernel<<<dim3(B_DIM / 32), 256, 0, stream>>>(query,  pWq,  bq,  gq,  cq,  qn,  nullptr, 0);
    proj_gn_kernel<<<dim3(B_DIM / 32), 256, 0, stream>>>(value1, pWv1, bv1, gv1, cv1, v1n, nullptr, 0);
    proj_gn_kernel<<<dim3(B_DIM * M_DIM / 32), 256, 0, stream>>>(key, pWk, bk, gk, ck,
                                                                 nullptr, kn, 1);
    // 3) scores: logits + masked pool (h never materialized)
    score_kernel<<<dim3(B_DIM * H_DIM), 256, 0, stream>>>(qn, kn, pW1, b1, Wls, bls, mask,
                                                          logits, pool);
    // 4) softmax
    softmax_kernel<<<dim3(B_DIM * H_DIM), 128, 0, stream>>>(logits, mask, alpha);
    // 5) fused v2 projection + aggregation
    hipMemsetAsync(v2agg, 0, (size_t)B_DIM * E_DIM * 4, stream);
    v2_proj_agg_kernel<<<dim3(B_DIM * M_DIM / 32), 256, 0, stream>>>(value2, pWv2, bv2, gv2, cv2,
                                                                     alpha, v2agg);
    // 6) channel gate + final product
    final_kernel<<<dim3((B_DIM * E_DIM + 255) / 256), 256, 0, stream>>>(v1n, v2agg, pool,
                                                                        Wlc, blc, out);
}
